// HungarianMatcher_45226005627158
// MI455X (gfx1250) — compile-verified, matrix operand order checked by reading, not run on hardware
//
#include <hip/hip_runtime.h>
#include <hip/hip_bf16.h>
#include <math.h>

typedef __attribute__((ext_vector_type(16))) _Float16 v16h;
typedef __attribute__((ext_vector_type(8)))  float    v8f;
typedef __attribute__((address_space(3))) float lds_float;

#define BATCH 8
#define NPRED 1024
#define NCLS  4
#define NGT   64
#define NP    20
#define HN    64      // Hungarian rows (gts)
#define HM    1024    // Hungarian cols (preds)
#define BIG   3.0e38f

// ---------------------------------------------------------------------------
// 1) softmax over the 4 class logits
// ---------------------------------------------------------------------------
__global__ void softmax4_kernel(const float* __restrict__ logits,
                                float* __restrict__ probs) {
    int t = blockIdx.x * blockDim.x + threadIdx.x;   // (b*1024 + i)
    if (t >= BATCH * NPRED) return;
    const float* lg = logits + (size_t)t * NCLS;
    float a0 = lg[0], a1 = lg[1], a2 = lg[2], a3 = lg[3];
    float m  = fmaxf(fmaxf(a0, a1), fmaxf(a2, a3));
    float e0 = expf(a0 - m), e1 = expf(a1 - m), e2 = expf(a2 - m), e3 = expf(a3 - m);
    float r  = 1.0f / (e0 + e1 + e2 + e3);
    float* pr = probs + (size_t)t * NCLS;
    pr[0] = e0 * r; pr[1] = e1 * r; pr[2] = e2 * r; pr[3] = e3 * r;
}

// ---------------------------------------------------------------------------
// 2) masked-L1 point cost cp[b][i][j]
//    gt geometry staged into LDS with CDNA5 async-to-LDS B128 transfers
//    (no VGPR staging), completion via s_wait_asynccnt before the barrier.
// ---------------------------------------------------------------------------
#define SGX(j, p) sgl[(((j) * NP + (p)) << 1) + 0]
#define SGY(j, p) sgl[(((j) * NP + (p)) << 1) + 1]
#define SMK(j, p) smsk[(j) * NP + (p)]

__global__ void cp_kernel(const float* __restrict__ pred_lines,
                          const float* __restrict__ gt_lines,
                          const int*   __restrict__ gt_labels,
                          const float* __restrict__ gt_masks,
                          float* __restrict__ cp) {
    __shared__ __align__(16) float sgl[NGT * NP * 2];   // interleaved x,y (10 KB)
    __shared__ __align__(16) float smsk[NGT * NP];      // 5 KB
    __shared__ float snv[NGT];
    __shared__ int   slb[NGT];

    int b   = blockIdx.x >> 2;                 // 4 blocks per batch
    int i   = ((blockIdx.x & 3) << 8) + threadIdx.x;
    int tid = threadIdx.x;

    // LDS byte offsets for async destinations (generic -> AS3 -> offset)
    unsigned lds_gl = (unsigned)(unsigned long long)(lds_float*)&sgl[0];
    unsigned lds_mk = (unsigned)(unsigned long long)(lds_float*)&smsk[0];

    // gt_lines slab for this batch: 2560 floats = 640 x B128
    {
        unsigned long long src0 =
            (unsigned long long)(const void*)(gt_lines + (size_t)b * NGT * NP * 2);
        for (int k = tid; k < (NGT * NP * 2) / 4; k += 256) {
            unsigned           dst = lds_gl + (unsigned)k * 16u;
            unsigned long long src = src0 + (unsigned long long)k * 16ull;
            asm volatile("global_load_async_to_lds_b128 %0, %1, off"
                         :: "v"(dst), "v"(src) : "memory");
        }
    }
    // gt_masks slab: 1280 floats = 320 x B128
    {
        unsigned long long src0 =
            (unsigned long long)(const void*)(gt_masks + (size_t)b * NGT * NP);
        for (int k = tid; k < (NGT * NP) / 4; k += 256) {
            unsigned           dst = lds_mk + (unsigned)k * 16u;
            unsigned long long src = src0 + (unsigned long long)k * 16ull;
            asm volatile("global_load_async_to_lds_b128 %0, %1, off"
                         :: "v"(dst), "v"(src) : "memory");
        }
    }
    asm volatile("s_wait_asynccnt 0" ::: "memory");
    __syncthreads();

    if (tid < NGT) {
        float s = 0.0f;
        for (int p = 0; p < NP; p++) s += SMK(tid, p);
        snv[tid] = fmaxf(s, 1.0f);
        slb[tid] = gt_labels[b * NGT + tid];
    }
    __syncthreads();

    const float* pl = pred_lines + (size_t)(b * NPRED + i) * NP * 2;
    __builtin_prefetch(pl, 0, 0);              // global_prefetch_b8
    float px[NP], py[NP];
    for (int p = 0; p < NP; p++) { px[p] = pl[2 * p]; py[p] = pl[2 * p + 1]; }

    float* out = cp + (size_t)b * NPRED * NGT + (size_t)i * NGT;
    for (int j = 0; j < NGT; j++) {
        int   lbl = slb[j];
        float c;
        if (lbl == 1) {                        // polygon: min over cyclic shifts
            float best = BIG;
            for (int s = 0; s < NP; s++) {
                float acc = 0.0f;
                for (int p = 0; p < NP; p++) {
                    int q = p + s; if (q >= NP) q -= NP;
                    acc += SMK(j, q) * (fabsf(px[p] - SGX(j, q)) + fabsf(py[p] - SGY(j, q)));
                }
                best = fminf(best, acc);
            }
            c = best / snv[j];
        } else if (lbl == 0 || lbl == 2) {     // line: forward vs backward
            float cf = 0.0f, cb = 0.0f;
            for (int p = 0; p < NP; p++) {
                cf += SMK(j, p) * (fabsf(px[p] - SGX(j, p)) + fabsf(py[p] - SGY(j, p)));
                int q = NP - 1 - p;
                cb += SMK(j, q) * (fabsf(px[p] - SGX(j, q)) + fabsf(py[p] - SGY(j, q)));
            }
            c = fminf(cf, cb) / snv[j];
        } else {
            c = 0.0f;
        }
        out[j] = c;
    }
}

// ---------------------------------------------------------------------------
// 3) class cost via WMMA (probs x scaled-onehot, K=4 padded into K=32),
//    fused with 5*cp, nan_to_num, written transposed (G x N) for Hungarian.
//    One wave handles one 16-row M tile x 4 N tiles.
// ---------------------------------------------------------------------------
__global__ void combine_wmma_kernel(const float* __restrict__ probs,
                                    const int*   __restrict__ gt_labels,
                                    const float* __restrict__ cp,
                                    float* __restrict__ cost) {
    int gtid = blockIdx.x * blockDim.x + threadIdx.x;
    int wave = gtid >> 5;                      // 512 waves total
    int lane = threadIdx.x & 31;
    int b    = wave >> 6;                      // 64 M tiles per batch
    int mt   = wave & 63;
    int m0   = mt * 16;

    // A tile (16x32 f16): lanes 0-15, elements 0..3 = probs[m0+lane][0..3]; rest 0
    v16h a = {};
    if (lane < 16) {
        const float* pr = probs + (size_t)(b * NPRED + m0 + lane) * NCLS;
        a[0] = (_Float16)pr[0];
        a[1] = (_Float16)pr[1];
        a[2] = (_Float16)pr[2];
        a[3] = (_Float16)pr[3];
    }

    for (int nt = 0; nt < 4; nt++) {
        // B tile (32x16 f16): lane n<16 holds column n; element k = (label==k)?-2:0
        v16h bb = {};
        if (lane < 16) {
            int lbl = gt_labels[b * NGT + nt * 16 + lane];
            const _Float16 w = (_Float16)(-2.0f);   // -COST_CLASS
            bb[0] = (lbl == 0) ? w : (_Float16)0.0f;
            bb[1] = (lbl == 1) ? w : (_Float16)0.0f;
            bb[2] = (lbl == 2) ? w : (_Float16)0.0f;
            bb[3] = (lbl == 3) ? w : (_Float16)0.0f;
        }
        v8f c = {};
        c = __builtin_amdgcn_wmma_f32_16x16x32_f16(false, a, false, bb,
                                                   (short)0, c, false, false);
        // C/D layout: reg r -> (M = r (+8 for hi lanes), N = lane&15)
        int n     = nt * 16 + (lane & 15);
        int mbase = m0 + ((lane < 16) ? 0 : 8);
        for (int r = 0; r < 8; r++) {
            int   m   = mbase + r;
            float val = c[r] + 5.0f * cp[(size_t)(b * NPRED + m) * NGT + n];
            if (!(val == val))      val = 1.0e6f;     // NaN
            else if (val >  BIG)    val = 1.0e6f;     // +inf
            else if (val < -BIG)    val = -1.0e6f;    // -inf
            cost[(size_t)b * NGT * NPRED + (size_t)n * NPRED + m] = val;
        }
    }
}

// ---------------------------------------------------------------------------
// 4) exact rectangular Hungarian (JV augmenting paths), one block per batch.
//    Sequential over rows/paths; O(N) scans parallelized across 256 threads.
// ---------------------------------------------------------------------------
__global__ void hungarian_kernel(const float* __restrict__ cost,
                                 float* __restrict__ out_idx,   // d_out[0..1023]
                                 int* __restrict__ pred_i,
                                 int* __restrict__ gt_i) {
    __shared__ float su[HN + 1];
    __shared__ float sv[HM + 1];
    __shared__ int   sp[HM + 1];
    __shared__ int   sway[HM + 1];
    __shared__ float sminv[HM + 1];
    __shared__ int   sused[HM + 1];
    __shared__ int   scol[HN];
    __shared__ float rmin[256];
    __shared__ int   rjdx[256];

    int b   = blockIdx.x;
    int tid = threadIdx.x;
    const float* C = cost + (size_t)b * HN * HM;

    for (int j = tid; j <= HM; j += 256) { sv[j] = 0.0f; sp[j] = 0; sway[j] = 0; }
    for (int j = tid; j <= HN; j += 256) su[j] = 0.0f;
    __syncthreads();

    for (int i = 1; i <= HN; i++) {
        if (tid == 0) sp[0] = i;
        for (int j = tid; j <= HM; j += 256) { sminv[j] = BIG; sused[j] = 0; }
        __syncthreads();

        int j0 = 0;
        while (true) {
            if (tid == 0) sused[j0] = 1;
            __syncthreads();
            int   i0  = sp[j0];
            float ui0 = su[i0];
            for (int j = 1 + tid; j <= HM; j += 256) {
                if (!sused[j]) {
                    float cur = C[(size_t)(i0 - 1) * HM + (j - 1)] - ui0 - sv[j];
                    if (cur < sminv[j]) { sminv[j] = cur; sway[j] = j0; }
                }
            }
            __syncthreads();
            // argmin over free columns
            float best = BIG; int bj = -1;
            for (int j = 1 + tid; j <= HM; j += 256) {
                if (!sused[j] && sminv[j] < best) { best = sminv[j]; bj = j; }
            }
            rmin[tid] = best; rjdx[tid] = bj;
            __syncthreads();
            for (int off = 128; off > 0; off >>= 1) {
                if (tid < off) {
                    float o = rmin[tid + off]; int oj = rjdx[tid + off];
                    if (oj != -1 && (o < rmin[tid] ||
                        (o == rmin[tid] && (rjdx[tid] == -1 || oj < rjdx[tid])))) {
                        rmin[tid] = o; rjdx[tid] = oj;
                    }
                }
                __syncthreads();
            }
            float delta = rmin[0];
            int   j1    = rjdx[0];
            __syncthreads();
            // dual updates (p[j] distinct for used j -> conflict-free)
            for (int j = tid; j <= HM; j += 256) {
                if (sused[j]) { su[sp[j]] += delta; sv[j] -= delta; }
                else          { sminv[j] -= delta; }
            }
            __syncthreads();
            j0 = j1;
            if (sp[j0] == 0) break;
        }
        // augment (serial)
        if (tid == 0) {
            int jj = j0;
            while (jj != 0) { int jp = sway[jj]; sp[jj] = sp[jp]; jj = jp; }
        }
        __syncthreads();
    }

    // extract col4row
    for (int j = 1 + tid; j <= HM; j += 256) {
        int pj = sp[j];
        if (pj > 0) scol[pj - 1] = j - 1;
    }
    __syncthreads();
    // argsort by assigned column (all distinct) -> rank via counting
    for (int g = tid; g < HN; g += 256) {
        int cg = scol[g];
        int rank = 0;
        for (int h = 0; h < HN; h++) rank += (scol[h] < cg) ? 1 : 0;
        out_idx[b * NGT + rank]               = (float)cg;   // pred_idx
        out_idx[BATCH * NGT + b * NGT + rank] = (float)g;    // gt_idx
        pred_i[b * NGT + rank] = cg;
        gt_i[b * NGT + rank]   = g;
    }
}

// ---------------------------------------------------------------------------
// 5) gather matched pairs, pick best gt orientation / cyclic shift
// ---------------------------------------------------------------------------
__global__ void best_kernel(const float* __restrict__ pred_lines,
                            const float* __restrict__ gt_lines,
                            const int*   __restrict__ gt_labels,
                            const float* __restrict__ gt_masks,
                            const int*   __restrict__ pred_i,
                            const int*   __restrict__ gt_i,
                            float* __restrict__ out_best) {
    int t = blockIdx.x * blockDim.x + threadIdx.x;
    if (t >= BATCH * NGT) return;
    int b  = t >> 6;
    int pi = pred_i[t];
    int gi = gt_i[t];

    const float* pl = pred_lines + (size_t)(b * NPRED + pi) * NP * 2;
    const float* gl = gt_lines   + (size_t)(b * NGT  + gi) * NP * 2;
    const float* ml = gt_masks   + (size_t)(b * NGT  + gi) * NP;
    int lbl = gt_labels[b * NGT + gi];

    float px[NP], py[NP], gx[NP], gy[NP], mk[NP];
    for (int p = 0; p < NP; p++) {
        px[p] = pl[2 * p]; py[p] = pl[2 * p + 1];
        gx[p] = gl[2 * p]; gy[p] = gl[2 * p + 1];
        mk[p] = ml[p];
    }

    float cf = 0.0f, cb = 0.0f;
    for (int p = 0; p < NP; p++) {
        cf += mk[p] * (fabsf(px[p] - gx[p]) + fabsf(py[p] - gy[p]));
        int q = NP - 1 - p;
        cb += mk[q] * (fabsf(px[p] - gx[q]) + fabsf(py[p] - gy[q]));
    }
    float bestc = BIG; int bs = 0;
    for (int s = 0; s < NP; s++) {
        float acc = 0.0f;
        for (int p = 0; p < NP; p++) {
            int q = p + s; if (q >= NP) q -= NP;
            acc += mk[q] * (fabsf(px[p] - gx[q]) + fabsf(py[p] - gy[q]));
        }
        if (acc < bestc) { bestc = acc; bs = s; }
    }

    float* o = out_best + (size_t)t * NP * 2;
    if (lbl == 1) {                      // polygon: best cyclic shift
        for (int p = 0; p < NP; p++) {
            int q = p + bs; if (q >= NP) q -= NP;
            o[2 * p] = gx[q]; o[2 * p + 1] = gy[q];
        }
    } else if (lbl == 0 || lbl == 2) {   // line: best direction
        if (cf < cb) {
            for (int p = 0; p < NP; p++) { o[2 * p] = gx[p]; o[2 * p + 1] = gy[p]; }
        } else {
            for (int p = 0; p < NP; p++) {
                int q = NP - 1 - p;
                o[2 * p] = gx[q]; o[2 * p + 1] = gy[q];
            }
        }
    } else {
        for (int p = 0; p < NP; p++) { o[2 * p] = gx[p]; o[2 * p + 1] = gy[p]; }
    }
}

// ---------------------------------------------------------------------------
extern "C" void kernel_launch(void* const* d_in, const int* in_sizes, int n_in,
                              void* d_out, int out_size, void* d_ws, size_t ws_size,
                              hipStream_t stream) {
    const float* logits     = (const float*)d_in[0];   // (8,1024,4)
    const float* pred_lines = (const float*)d_in[1];   // (8,1024,20,2)
    const int*   gt_labels  = (const int*)  d_in[2];   // (8,64)
    const float* gt_lines   = (const float*)d_in[3];   // (8,64,20,2)
    const float* gt_masks   = (const float*)d_in[4];   // (8,64,20)
    float* out = (float*)d_out;                        // 512 + 512 + 20480 floats

    char*  ws    = (char*)d_ws;
    float* probs = (float*)ws;                                   // 32768 f
    float* cp    = (float*)(ws + 131072);                        // 524288 f
    float* cost  = (float*)(ws + 131072 + 2097152);              // 524288 f
    int*   predi = (int*)  (ws + 131072 + 2 * 2097152);          // 512 i
    int*   gti   = predi + BATCH * NGT;                          // 512 i

    softmax4_kernel   <<<32, 256, 0, stream>>>(logits, probs);
    cp_kernel         <<<BATCH * 4, 256, 0, stream>>>(pred_lines, gt_lines,
                                                      gt_labels, gt_masks, cp);
    combine_wmma_kernel<<<64, 256, 0, stream>>>(probs, gt_labels, cp, cost);
    hungarian_kernel  <<<BATCH, 256, 0, stream>>>(cost, out, predi, gti);
    best_kernel       <<<2, 256, 0, stream>>>(pred_lines, gt_lines, gt_labels,
                                              gt_masks, predi, gti,
                                              out + 2 * BATCH * NGT);
}
